// HAN_88888643158475
// MI455X (gfx1250) — compile-verified, matrix operand order checked by reading, not run on hardware
//
#include <hip/hip_runtime.h>

typedef _Float16 f16;
typedef __attribute__((ext_vector_type(16))) _Float16 v16h;
typedef __attribute__((ext_vector_type(8)))  float    v8f;

#define DV 128   // D_IN == D_OUT == H_SEM == 128
#define MP 3     // meta-paths

#define GRID_STRIDE(i, total) \
  for (long long i = (long long)blockIdx.x * blockDim.x + threadIdx.x; \
       i < (long long)(total); i += (long long)gridDim.x * blockDim.x)

// ---------------------------------------------------------------- helpers

__device__ inline v8f wmma16x16x32(v16h a, v16h b, v8f c) {
  // (neg_a, A, neg_b, B, c_mod, C, reuse_a, reuse_b)
  return __builtin_amdgcn_wmma_f32_16x16x32_f16(false, a, false, b, (short)0, c,
                                                false, false);
}

// 16-bit A-matrix 16x32 fragment layout (CDNA5 ISA 7.12.2):
// lane L holds row M = L&15; K-halves: [kb + 8*(L>>4) + 0..7] and [+16..+23].
__device__ inline v16h frag_ld_f16(const f16* __restrict__ base, int ld, int kb,
                                   int lane) {
  const f16* p = base + (long long)(lane & 15) * ld + kb + ((lane >> 4) << 3);
  v16h f;
#pragma unroll
  for (int i = 0; i < 8; ++i) { f[i] = p[i]; f[8 + i] = p[16 + i]; }
  return f;
}

__device__ inline v16h frag_ld_f32(const float* __restrict__ base, int ld, int kb,
                                   int lane) {
  const float* p = base + (long long)(lane & 15) * ld + kb + ((lane >> 4) << 3);
  v16h f;
#pragma unroll
  for (int i = 0; i < 8; ++i) { f[i] = (f16)p[i]; f[8 + i] = (f16)p[16 + i]; }
  return f;
}

// monotonic uint encoding of float for atomicMax-based segment_max
__device__ inline unsigned f2mono(float f) {
  unsigned u = __float_as_uint(f);
  return (u >> 31) ? ~u : (u | 0x80000000u);
}
__device__ inline float mono2f(unsigned u) {
  return __uint_as_float((u >> 31) ? (u & 0x7FFFFFFFu) : ~u);
}

__device__ inline float lrelu(float x) { return x > 0.f ? x : 0.2f * x; }

// ---------------------------------------------------------------- kernels

// f32 -> f16 conversions + transposes for WMMA B operands
__global__ void k_convert(const float* __restrict__ h, const float* __restrict__ W,
                          const float* __restrict__ Ws1, f16* __restrict__ h16,
                          f16* __restrict__ Wt16, f16* __restrict__ Ws1t16, int N) {
  GRID_STRIDE(i, (long long)N * DV) h16[i] = (f16)h[i];
  GRID_STRIDE(i, MP * DV * DV) {   // Wt16[m][o][k] = W[m][k][o]
    int m = (int)(i / (DV * DV)), r = (int)(i % (DV * DV));
    int o = r / DV, k = r % DV;
    Wt16[i] = (f16)W[((long long)m * DV + k) * DV + o];
  }
  GRID_STRIDE(i, DV * DV) {        // Ws1t16[hh][d] = Ws1[d][hh]
    int hh = (int)(i / DV), d = (int)(i % DV);
    Ws1t16[i] = (f16)Ws1[d * DV + hh];
  }
}

__global__ void k_init(float* __restrict__ z, unsigned* __restrict__ emax,
                       float* __restrict__ denom, float* __restrict__ wsum, int N) {
  GRID_STRIDE(i, (long long)MP * N * DV) z[i] = 0.f;
  GRID_STRIDE(i, (long long)MP * N) { emax[i] = 0x007FFFFFu /*mono(-inf)*/; denom[i] = 0.f; }
  GRID_STRIDE(i, 4) wsum[i] = 0.f;
}

// feat16[m] = h @ W[m]   (block = 8 waves = 16 rows x 128 cols; wave = 16x16 tile)
__global__ void k_gemm_feat(const f16* __restrict__ h16, const f16* __restrict__ Wt16,
                            f16* __restrict__ feat16, int N) {
  const int m = blockIdx.y;
  const int lane = threadIdx.x & 31;
  const int wave = threadIdx.x >> 5;      // col tile 0..7
  const long long rowbase = (long long)blockIdx.x * 16;
  const f16* A = h16 + rowbase * DV;
  const f16* B = Wt16 + (long long)(m * DV + wave * 16) * DV;
  v8f c = {};
#pragma unroll
  for (int kb = 0; kb < DV; kb += 32) {
    v16h a = frag_ld_f16(A, DV, kb, lane);
    v16h b = frag_ld_f16(B, DV, kb, lane);
    c = wmma16x16x32(a, b, c);
  }
  const int col  = wave * 16 + (lane & 15);
  const int rsub = (lane >> 4) * 8;
  f16* out = feat16 + ((long long)m * N + rowbase) * DV;
#pragma unroll
  for (int r = 0; r < 8; ++r)
    out[(long long)(rsub + r) * DV + col] = (f16)c[r];
}

// el/er: per-(m,n) dot of feat row with attention vectors (one wave per row)
__global__ void k_attn(const f16* __restrict__ feat16, const float* __restrict__ al,
                       const float* __restrict__ ar, float* __restrict__ el,
                       float* __restrict__ er, int N) {
  const int lane = threadIdx.x & 31;
  const long long row = ((long long)blockIdx.x * blockDim.x + threadIdx.x) >> 5;
  if (row >= (long long)MP * N) return;
  const int m = (int)(row / N);
  const f16* f = feat16 + row * DV;
  float sl = 0.f, sr = 0.f;
#pragma unroll
  for (int j = 0; j < 4; ++j) {
    int o = lane * 4 + j;
    float v = (float)f[o];
    sl += v * al[m * DV + o];
    sr += v * ar[m * DV + o];
  }
  for (int off = 16; off; off >>= 1) { sl += __shfl_xor(sl, off); sr += __shfl_xor(sr, off); }
  if (lane == 0) { el[row] = sl; er[row] = sr; }
}

__global__ void k_edge_max(const int* __restrict__ src, const int* __restrict__ dst,
                           const float* __restrict__ el, const float* __restrict__ er,
                           unsigned* __restrict__ emax, int E, int N) {
  long long i = (long long)blockIdx.x * blockDim.x + threadIdx.x;
  if (i >= (long long)MP * E) return;
  int m = (int)(i / E);
  float v = lrelu(el[(long long)m * N + src[i]] + er[(long long)m * N + dst[i]]);
  atomicMax(&emax[(long long)m * N + dst[i]], f2mono(v));
}

__global__ void k_edge_sum(const int* __restrict__ src, const int* __restrict__ dst,
                           const float* __restrict__ el, const float* __restrict__ er,
                           const unsigned* __restrict__ emax, float* __restrict__ denom,
                           int E, int N) {
  long long i = (long long)blockIdx.x * blockDim.x + threadIdx.x;
  if (i >= (long long)MP * E) return;
  int m = (int)(i / E);
  long long nd = (long long)m * N + dst[i];
  float v = lrelu(el[(long long)m * N + src[i]] + er[nd]);
  atomicAdd(&denom[nd], expf(v - mono2f(emax[nd])));
}

// one wave per edge: z[dst] += a * feat[src]   (4 f32 atomics per lane)
__global__ void k_edge_agg(const int* __restrict__ src, const int* __restrict__ dst,
                           const float* __restrict__ el, const float* __restrict__ er,
                           const unsigned* __restrict__ emax, const float* __restrict__ denom,
                           const f16* __restrict__ feat16, float* __restrict__ z,
                           int E, int N) {
  const int lane = threadIdx.x & 31;
  const long long w = ((long long)blockIdx.x * blockDim.x + threadIdx.x) >> 5;
  if (w >= (long long)MP * E) return;
  const int m = (int)(w / E);
  const long long ns = (long long)m * N + src[w];
  const long long nd = (long long)m * N + dst[w];
  const float v = lrelu(el[ns] + er[nd]);
  const float a = expf(v - mono2f(emax[nd])) / denom[nd];
  const f16* fs = feat16 + ns * DV + lane * 4;
  float* zd = z + nd * DV + lane * 4;
#pragma unroll
  for (int j = 0; j < 4; ++j) atomicAdd(&zd[j], a * (float)fs[j]);
}

__global__ void k_bias_elu(float* __restrict__ z, const float* __restrict__ bias, int N) {
  GRID_STRIDE(i, (long long)MP * N * DV) {
    int m = (int)(i / ((long long)N * DV));
    int o = (int)(i % DV);
    float x = z[i] + bias[m * DV + o];
    z[i] = x > 0.f ? x : expm1f(x);
  }
}

// wsum[m] += sum_n Ws2 . tanh(z[m,n,:] @ Ws1 + bs1)   (wave = 16 nodes, WMMA)
__global__ void k_semantic(const float* __restrict__ z, const f16* __restrict__ Ws1t16,
                           const float* __restrict__ bs1, const float* __restrict__ Ws2,
                           float* __restrict__ wsum, int N) {
  const int m = blockIdx.y;
  const int lane = threadIdx.x & 31;
  const int wave = threadIdx.x >> 5;
  const int rowtile = blockIdx.x * 8 + wave;       // wave-uniform guard -> EXEC all-1
  if (rowtile * 16 >= N) return;                   // N % 16 == 0 here
  const float* A = z + ((long long)m * N + (long long)rowtile * 16) * DV;

  v16h afrag[4];
#pragma unroll
  for (int ki = 0; ki < 4; ++ki) afrag[ki] = frag_ld_f32(A, DV, ki * 32, lane);

  const int col0 = lane & 15;
  float acc[8];
#pragma unroll
  for (int r = 0; r < 8; ++r) acc[r] = 0.f;

#pragma unroll
  for (int ct = 0; ct < 8; ++ct) {
    v8f c = {};
#pragma unroll
    for (int ki = 0; ki < 4; ++ki) {
      v16h b = frag_ld_f16(Ws1t16 + (long long)(ct * 16) * DV, DV, ki * 32, lane);
      c = wmma16x16x32(afrag[ki], b, c);
    }
    const int col = ct * 16 + col0;
    const float b1 = bs1[col], w2 = Ws2[col];
#pragma unroll
    for (int r = 0; r < 8; ++r) acc[r] += tanhf(c[r] + b1) * w2;
  }
  float s = 0.f;
#pragma unroll
  for (int r = 0; r < 8; ++r) s += acc[r];
  for (int off = 1; off < 16; off <<= 1) s += __shfl_xor(s, off, 16);
  if (col0 == 0) atomicAdd(&wsum[m], s);   // lanes 0 (rows 0-7) and 16 (rows 8-15)
}

__global__ void k_beta(const float* __restrict__ wsum, float* __restrict__ beta, int N) {
  if (threadIdx.x == 0 && blockIdx.x == 0) {
    float a0 = wsum[0] / N, a1 = wsum[1] / N, a2 = wsum[2] / N;
    float mx = fmaxf(a0, fmaxf(a1, a2));
    float e0 = expf(a0 - mx), e1 = expf(a1 - mx), e2 = expf(a2 - mx);
    float s = e0 + e1 + e2;
    beta[0] = e0 / s; beta[1] = e1 / s; beta[2] = e2 / s;
  }
}

__global__ void k_out(const float* __restrict__ z, const float* __restrict__ beta,
                      float* __restrict__ out, int N) {
  const float b0 = beta[0], b1 = beta[1], b2 = beta[2];
  const long long nd = (long long)N * DV;
  GRID_STRIDE(i, nd) out[i] = b0 * z[i] + b1 * z[nd + i] + b2 * z[2 * nd + i];
}

// ---------------------------------------------------------------- launch

extern "C" void kernel_launch(void* const* d_in, const int* in_sizes, int n_in,
                              void* d_out, int out_size, void* d_ws, size_t ws_size,
                              hipStream_t stream) {
  (void)n_in; (void)out_size; (void)ws_size;
  const float* h      = (const float*)d_in[0];
  const int*   esrc   = (const int*)d_in[1];
  const int*   edst   = (const int*)d_in[2];
  const float* W_gat  = (const float*)d_in[3];
  const float* attn_l = (const float*)d_in[4];
  const float* attn_r = (const float*)d_in[5];
  const float* bias   = (const float*)d_in[6];
  const float* Ws1    = (const float*)d_in[7];
  const float* bs1    = (const float*)d_in[8];
  const float* Ws2    = (const float*)d_in[9];
  float* out = (float*)d_out;

  const int N = in_sizes[0] / DV;   // 50000 (multiple of 16)
  const int E = in_sizes[1] / MP;   // 650000 edges per meta-path (incl. self-loops)

  char* ws = (char*)d_ws;
  size_t off = 0;
  auto alloc = [&](size_t bytes) -> void* {
    off = (off + 255) & ~(size_t)255;
    void* p = ws + off;
    off += bytes;
    return p;
  };
  f16*      h16    = (f16*)alloc((size_t)N * DV * sizeof(f16));
  f16*      Wt16   = (f16*)alloc((size_t)MP * DV * DV * sizeof(f16));
  f16*      Ws1t16 = (f16*)alloc((size_t)DV * DV * sizeof(f16));
  f16*      feat16 = (f16*)alloc((size_t)MP * N * DV * sizeof(f16));
  float*    el     = (float*)alloc((size_t)MP * N * sizeof(float));
  float*    er     = (float*)alloc((size_t)MP * N * sizeof(float));
  unsigned* emax   = (unsigned*)alloc((size_t)MP * N * sizeof(unsigned));
  float*    denom  = (float*)alloc((size_t)MP * N * sizeof(float));
  float*    z      = (float*)alloc((size_t)MP * N * DV * sizeof(float));
  float*    wsum   = (float*)alloc(16 * sizeof(float));
  float*    beta   = (float*)alloc(16 * sizeof(float));

  const int thr = 256;
  k_convert<<<4096, thr, 0, stream>>>(h, W_gat, Ws1, h16, Wt16, Ws1t16, N);
  k_init<<<8192, thr, 0, stream>>>(z, emax, denom, wsum, N);
  k_gemm_feat<<<dim3(N / 16, MP), 256, 0, stream>>>(h16, Wt16, feat16, N);

  const long long rows = (long long)MP * N;
  k_attn<<<(unsigned)((rows * 32 + thr - 1) / thr), thr, 0, stream>>>(
      feat16, attn_l, attn_r, el, er, N);

  const long long edges = (long long)MP * E;
  k_edge_max<<<(unsigned)((edges + thr - 1) / thr), thr, 0, stream>>>(
      esrc, edst, el, er, emax, E, N);
  k_edge_sum<<<(unsigned)((edges + thr - 1) / thr), thr, 0, stream>>>(
      esrc, edst, el, er, emax, denom, E, N);
  k_edge_agg<<<(unsigned)((edges * 32 + thr - 1) / thr), thr, 0, stream>>>(
      esrc, edst, el, er, emax, denom, feat16, z, E, N);

  k_bias_elu<<<8192, thr, 0, stream>>>(z, bias, N);
  k_semantic<<<dim3((N / 16 + 7) / 8, MP), 256, 0, stream>>>(z, Ws1t16, bs1, Ws2, wsum, N);
  k_beta<<<1, 32, 0, stream>>>(wsum, beta, N);
  k_out<<<4096, thr, 0, stream>>>(z, beta, out, N);
}
